// RelationContextEncoder_27127013441944
// MI455X (gfx1250) — matrix operand order repaired and checked by reading, not compiled, not run on hardware
//
#include <hip/hip_runtime.h>
#include <hip/hip_bf16.h>

typedef __attribute__((ext_vector_type(16))) _Float16 v16h;
typedef __attribute__((ext_vector_type(4)))  _Float16 v4h;
typedef __attribute__((ext_vector_type(8)))  float    v8f;

#define NUM_ENT 64
#define DIM_ENT 8
#define CTXW    (NUM_ENT * DIM_ENT)   // 512
#define NHID    64
#define NREL    2016
#define NOUT    256
#define BATCH   1024
#define MTILE   16
#define NKB     (NREL * NHID / 16)    // 8064 K-blocks of 16 rows
#define CSTRIDE 516                   // padded f32 row stride for ctx tile (bank-conflict free)
#define HSTRIDE 72                    // padded f16 row stride for h tile (16B-aligned, spread banks)

// ---------------------------------------------------------------------------
// Pack W_fc (f32, row-major [K=129024][256]) into WMMA-B fragment order, f16:
//   Bt[(kb*256 + n)*16 + q] = f16( W_fc[(kb*16 + q)*256 + n] )
// so each lane's 16-element B column-segment is one contiguous 32B v16h load.
// ---------------------------------------------------------------------------
__global__ __launch_bounds__(256)
void rce_pack_b(const float* __restrict__ W_fc, _Float16* __restrict__ Bt)
{
    const int kb = blockIdx.x;        // 0..NKB-1
    const int n  = threadIdx.x;       // 0..255
    const float* src = W_fc + (size_t)kb * 16 * NOUT + n;
    v16h o;
    #pragma unroll
    for (int q = 0; q < 16; ++q)
        o[q] = (_Float16)src[q * NOUT];
    ((v16h*)Bt)[(size_t)kb * NOUT + n] = o;
}

// ---------------------------------------------------------------------------
// Fused: stage-1 relation MLP (register-resident W_rel) -> stage-2 WMMA GEMM.
// grid = 64 M-tiles * kgroups K-splits; block = 256 threads = 8 waves.
// Each wave owns 2 N-tiles of 16 cols (8 waves * 32 = 256 output cols).
// B fragments for pair r are preloaded BEFORE stage-1 + barrier so their L2
// latency is hidden behind ~80 VALU ops and the barrier wait.
// ---------------------------------------------------------------------------
template<bool PACKED>
__global__ __launch_bounds__(256)
void rce_main(const float* __restrict__ ctx,
              const float* __restrict__ W_rel,
              const float* __restrict__ b_rel,
              const float* __restrict__ W_fc,
              const _Float16* __restrict__ Bt,
              const float* __restrict__ b_fc,
              float* __restrict__ out,      // used when kgroups == 1 (fused epilogue)
              float* __restrict__ part,     // used when kgroups  > 1 (fp32 partials)
              int kgroups, int ppg)
{
    __shared__ float    s_ctx[MTILE * CSTRIDE];      // ~33 KB
    __shared__ _Float16 s_h[2][MTILE * HSTRIDE];     // double-buffered h tile

    const int tid  = threadIdx.x;
    const int mt   = blockIdx.x % 64;     // batch M-tile
    const int g    = blockIdx.x / 64;     // K-split group
    const int lane = tid & 31;
    const int w    = tid >> 5;

    // stage-1 work mapping: thread -> (row, 4 hid units)
    const int row = tid & 15;
    const int hb  = (tid >> 4) << 2;      // 0..60 step 4

    // ---- stage ctx tile (16 x 512 f32) into LDS (coalesced b128, padded rows) ----
    {
        const float4* src = (const float4*)(ctx + (size_t)mt * MTILE * CTXW);
        #pragma unroll
        for (int q = 0; q < 8; ++q) {
            int p  = q * 256 + tid;        // float4 index 0..2047
            int rr = p >> 7;               // 128 float4 per row
            int c4 = p & 127;
            *(float4*)&s_ctx[rr * CSTRIDE + c4 * 4] = src[p];
        }
    }

    // ---- W_rel slice + bias: register-resident (no per-pair LDS re-reads) ----
    float  wreg[16][4];
    #pragma unroll
    for (int d = 0; d < 16; ++d) {
        float4 t = *(const float4*)&W_rel[d * NHID + hb];
        wreg[d][0] = t.x; wreg[d][1] = t.y; wreg[d][2] = t.z; wreg[d][3] = t.w;
    }
    const float4 bias4 = *(const float4*)&b_rel[hb];
    __syncthreads();

    // decode starting pair (i<j) for this K-group
    const int r0 = g * ppg, r1 = r0 + ppg;
    int ei = 0, rem = r0;
    while (rem >= (NUM_ENT - 1) - ei) { rem -= (NUM_ENT - 1) - ei; ++ei; }
    int ej = ei + 1 + rem;

    // WMMA lane mapping
    const int bM  = lane & 15;            // M (for A/C) and col-in-tile (for B/C)
    const int hiK = (lane >> 4) & 1;      // upper half-wave flag
    const int nb0 = w * 32;               // this wave's first N-tile col base

    v8f c0 = {}; v8f c1 = {};

    for (int r = r0; r < r1; ++r) {
        const int buf = r & 1;

        // ---------- preload all 4 B fragments for this pair (global, no h dep) ----------
        v16h Bf0, Bf1, Bf2, Bf3;
        if (PACKED) {
            const v16h* bt0 = (const v16h*)Bt + (size_t)(r * 4 + hiK) * NOUT;      // hid 0..31
            const v16h* bt1 = (const v16h*)Bt + (size_t)(r * 4 + 2 + hiK) * NOUT;  // hid 32..63
            Bf0 = bt0[nb0 + bM];
            Bf1 = bt0[nb0 + 16 + bM];
            Bf2 = bt1[nb0 + bM];
            Bf3 = bt1[nb0 + 16 + bM];
        } else {
            const float* bp0 = W_fc + ((size_t)r * NHID + (hiK ? 16 : 0)) * NOUT + nb0 + bM;
            const float* bp1 = bp0 + 32 * NOUT;
            #pragma unroll
            for (int q = 0; q < 16; ++q) {
                Bf0[q] = (_Float16)bp0[q * NOUT];
                Bf1[q] = (_Float16)bp0[q * NOUT + 16];
                Bf2[q] = (_Float16)bp1[q * NOUT];
                Bf3[q] = (_Float16)bp1[q * NOUT + 16];
            }
        }

        // ---------- stage 1: h[16x64] = relu(pair_feat @ W_rel + b_rel) -> f16 LDS ----------
        {
            const float* fi = &s_ctx[row * CSTRIDE + ei * DIM_ENT];
            const float* fj = &s_ctx[row * CSTRIDE + ej * DIM_ENT];
            float4 f0 = *(const float4*)fi;
            float4 f1 = *(const float4*)(fi + 4);
            float4 f2 = *(const float4*)fj;
            float4 f3 = *(const float4*)(fj + 4);
            float a0 = bias4.x, a1 = bias4.y, a2 = bias4.z, a3 = bias4.w;
            #define RCE_STEP(d, f) { a0 += (f) * wreg[d][0]; a1 += (f) * wreg[d][1]; \
                                     a2 += (f) * wreg[d][2]; a3 += (f) * wreg[d][3]; }
            RCE_STEP(0,  f0.x) RCE_STEP(1,  f0.y) RCE_STEP(2,  f0.z) RCE_STEP(3,  f0.w)
            RCE_STEP(4,  f1.x) RCE_STEP(5,  f1.y) RCE_STEP(6,  f1.z) RCE_STEP(7,  f1.w)
            RCE_STEP(8,  f2.x) RCE_STEP(9,  f2.y) RCE_STEP(10, f2.z) RCE_STEP(11, f2.w)
            RCE_STEP(12, f3.x) RCE_STEP(13, f3.y) RCE_STEP(14, f3.z) RCE_STEP(15, f3.w)
            #undef RCE_STEP
            v4h h;
            h[0] = (_Float16)fmaxf(a0, 0.f);
            h[1] = (_Float16)fmaxf(a1, 0.f);
            h[2] = (_Float16)fmaxf(a2, 0.f);
            h[3] = (_Float16)fmaxf(a3, 0.f);
            *(v4h*)&s_h[buf][row * HSTRIDE + hb] = h;
        }
        // prefetch next pair's B slab toward WGP caches (global_prefetch_b8)
        {
            int rn = (r + 1 < NREL) ? (r + 1) : r;
            if (PACKED)
                __builtin_prefetch(Bt + (size_t)rn * 4 * NOUT * 16 + tid * 64, 0, 3);
            else
                __builtin_prefetch(&W_fc[((size_t)rn * NHID + (tid >> 2)) * NOUT + (tid & 3) * 64], 0, 3);
        }
        __syncthreads();   // single barrier/iter; double buffer covers WAR hazard

        // ---------- stage 2: C[16x32 per wave] += h[16x64] @ W_fc[64x32] via WMMA ----------
        // A fragment (16x32 f16) from LDS, contiguous per-lane k order (same as B):
        // lanes 0-15: K kk+0..15 ; lanes 16-31: K kk+16..31
        {
            union { v16h v; float4 q[2]; } A;
            const _Float16* hp = &s_h[buf][bM * HSTRIDE + (hiK ? 16 : 0)];
            A.q[0] = *(const float4*)hp;
            A.q[1] = *(const float4*)(hp + 8);
            c0 = __builtin_amdgcn_wmma_f32_16x16x32_f16(false, A.v, false, Bf0, (short)0, c0, false, false);
            c1 = __builtin_amdgcn_wmma_f32_16x16x32_f16(false, A.v, false, Bf1, (short)0, c1, false, false);
            A.q[0] = *(const float4*)(hp + 32);
            A.q[1] = *(const float4*)(hp + 40);
            c0 = __builtin_amdgcn_wmma_f32_16x16x32_f16(false, A.v, false, Bf2, (short)0, c0, false, false);
            c1 = __builtin_amdgcn_wmma_f32_16x16x32_f16(false, A.v, false, Bf3, (short)0, c1, false, false);
        }

        if (++ej == NUM_ENT) { ++ei; ej = ei + 1; }
    }

    // ---------- epilogue ----------
    const int mrow = hiK ? 8 : 0;
    const int col  = nb0 + bM;
    if (kgroups == 1) {
        float bias0 = b_fc[col], bias1 = b_fc[col + 16];
        float* op = out + (size_t)mt * MTILE * NOUT;
        #pragma unroll
        for (int v = 0; v < 8; ++v) {
            op[(v + mrow) * NOUT + col]      = fmaxf(c0[v] + bias0, 0.f);
            op[(v + mrow) * NOUT + col + 16] = fmaxf(c1[v] + bias1, 0.f);
        }
    } else {
        float* pp = part + (size_t)(mt * kgroups + g) * MTILE * NOUT;
        #pragma unroll
        for (int v = 0; v < 8; ++v) {
            pp[(v + mrow) * NOUT + col]      = c0[v];
            pp[(v + mrow) * NOUT + col + 16] = c1[v];
        }
    }
}

// Sum K-split partials + bias + ReLU -> out [1024 x 256]
__global__ __launch_bounds__(256)
void rce_reduce(const float* __restrict__ part, const float* __restrict__ b_fc,
                float* __restrict__ out, int kgroups)
{
    int idx = blockIdx.x * 256 + threadIdx.x;   // 0..262143
    int b = idx >> 8, n = idx & 255;
    int mt = b >> 4, ml = b & 15;
    float s = b_fc[n];
    const float* pp = part + (size_t)mt * kgroups * MTILE * NOUT + ml * NOUT + n;
    for (int g = 0; g < kgroups; ++g)
        s += pp[(size_t)g * MTILE * NOUT];
    out[idx] = fmaxf(s, 0.f);
}

extern "C" void kernel_launch(void* const* d_in, const int* in_sizes, int n_in,
                              void* d_out, int out_size, void* d_ws, size_t ws_size,
                              hipStream_t stream)
{
    (void)in_sizes; (void)n_in; (void)out_size;
    const float* ctx   = (const float*)d_in[0];
    const float* W_rel = (const float*)d_in[1];
    const float* b_rel = (const float*)d_in[2];
    const float* W_fc  = (const float*)d_in[3];
    const float* b_fc  = (const float*)d_in[4];
    float* out = (float*)d_out;

    const size_t btBytes   = (size_t)NREL * NHID * NOUT * sizeof(_Float16); // ~66 MB
    const size_t partBytes = (size_t)64 * 8 * MTILE * NOUT * sizeof(float); // 8 MB

    if (ws_size >= btBytes + partBytes) {
        // packed path: pre-convert W_fc into WMMA-B fragment layout, then fused GEMM
        _Float16* Bt   = (_Float16*)d_ws;
        float*    part = (float*)((char*)d_ws + btBytes);
        int kgroups = 8, ppg = NREL / 8;
        hipLaunchKernelGGL(rce_pack_b, dim3(NKB), dim3(256), 0, stream, W_fc, Bt);
        hipLaunchKernelGGL((rce_main<true>), dim3(64 * kgroups), dim3(256), 0, stream,
                           ctx, W_rel, b_rel, W_fc, Bt, b_fc, out, part, kgroups, ppg);
        hipLaunchKernelGGL(rce_reduce, dim3(BATCH * NOUT / 256), dim3(256), 0, stream,
                           part, b_fc, out, kgroups);
    } else if (ws_size >= partBytes) {
        // unpacked K-split path
        float* part = (float*)d_ws;
        int kgroups = 8, ppg = NREL / 8;
        hipLaunchKernelGGL((rce_main<false>), dim3(64 * kgroups), dim3(256), 0, stream,
                           ctx, W_rel, b_rel, W_fc, (const _Float16*)nullptr, b_fc, out, part,
                           kgroups, ppg);
        hipLaunchKernelGGL(rce_reduce, dim3(BATCH * NOUT / 256), dim3(256), 0, stream,
                           part, b_fc, out, kgroups);
    } else {
        // minimal-ws fallback: single K-group, fused epilogue
        hipLaunchKernelGGL((rce_main<false>), dim3(64), dim3(256), 0, stream,
                           ctx, W_rel, b_rel, W_fc, (const _Float16*)nullptr, b_fc, out,
                           (float*)d_ws, 1, NREL);
    }
}
